// MMMO_32985348833662
// MI455X (gfx1250) — matrix operand order repaired and checked by reading, not compiled, not run on hardware
//
#include <hip/hip_runtime.h>
#include <hip/hip_bf16.h>

// ---------------------------------------------------------------------------
// MGAT multimodal GNN forward, MI455X (gfx1250, wave32).
// Dense GEMMs: v_wmma_f32_16x16x32_bf16, 128x128x32 LDS tiles, 8 waves,
// register-pipelined float4 staging with guarded fallback for edge tiles.
// bf16 conversion via native casts (v_cvt_pk_bf16_f32 path).
// ---------------------------------------------------------------------------

#define ACT_NONE  0
#define ACT_TANH  1
#define ACT_LEAKY 2

typedef __attribute__((ext_vector_type(16))) __bf16 v16bf;
typedef __attribute__((ext_vector_type(2)))  __bf16 bf16x2;
typedef __attribute__((ext_vector_type(8)))  float  v8f;

__device__ __forceinline__ unsigned short f2bf(float f) {
  return __builtin_bit_cast(unsigned short, (__bf16)f);
}
__device__ __forceinline__ unsigned pack2bf(float a, float b) {
  bf16x2 p;
  p.x = (__bf16)a;
  p.y = (__bf16)b;
  return __builtin_bit_cast(unsigned, p);
}

__device__ __forceinline__ float act_apply(float x, int act) {
  if (act == ACT_TANH)  return tanhf(x);
  if (act == ACT_LEAKY) return x > 0.f ? x : 0.01f * x;
  return x;
}

// ---------------------------------------------------------------------------
// WMMA GEMM: C[M,N] = epilogue(A[M,K] @ W[K,N]) ; row-major f32 operands,
// converted to bf16 while staging.  Epilogue:
//   addBeforeAct==0:  C = act(S + bias) (+ D)
//   addBeforeAct==1:  C = act(S + bias + D)
// ---------------------------------------------------------------------------
__global__ __launch_bounds__(256) void gemm_bf16_kernel(
    const float* __restrict__ A, const float* __restrict__ W,
    const float* __restrict__ bias, const float* __restrict__ D,
    float* __restrict__ C, int M, int N, int K, int act, int addBeforeAct)
{
  // LDS, swizzled so each lane's 16-element fragment is 32 contiguous bytes.
  // A element (row,k) -> As[row*32 + ((k>>3)&1)*16 + (k&7) + ((k>>4)<<3)]
  // W element (k,col) -> Bs[col*32 + ((k>>3)&1)*16 + (k&7) + ((k>>4)<<3)]
  __shared__ __align__(32) unsigned short As[128 * 32];
  __shared__ __align__(32) unsigned short Bs[128 * 32];

  const int tid   = threadIdx.x;
  const int lane  = tid & 31;
  const int wave  = tid >> 5;
  const int waveM = wave >> 1;        // 0..3
  const int waveN = wave & 1;         // 0..1
  const int half  = lane >> 4;
  const int l16   = lane & 15;
  const int tileM = blockIdx.y * 128;
  const int tileN = blockIdx.x * 128;

  const bool kAligned = (K & 3) == 0;
  const bool nAligned = (N & 3) == 0;
  const bool mFull = (tileM + 128 <= M);
  const bool nFull = (tileN + 128 <= N);

  v8f acc[2][4] = {};
  float4 ra[4], rb[4];   // register-pipelined staging buffers

  auto load_tiles = [&](int kk0) {
    const bool kf = (kk0 + 32 <= K);
    const bool fa = mFull && kf && kAligned;
    const bool fb = nFull && kf && nAligned && kAligned;
    for (int it = 0; it < 4; ++it) {
      int q = it * 256 + tid;                   // 0..1023 (1024 float4 per tile)
      {                                         // ---- A tile 128x32 ----
        int row = q >> 3, kk = (q & 7) * 4;
        int gr = tileM + row;
        if (fa) {
          ra[it] = *reinterpret_cast<const float4*>(A + (long)gr * K + kk0 + kk);
        } else {
          float4 t;
          t.x = (gr < M && kk0 + kk + 0 < K) ? A[(long)gr * K + kk0 + kk + 0] : 0.f;
          t.y = (gr < M && kk0 + kk + 1 < K) ? A[(long)gr * K + kk0 + kk + 1] : 0.f;
          t.z = (gr < M && kk0 + kk + 2 < K) ? A[(long)gr * K + kk0 + kk + 2] : 0.f;
          t.w = (gr < M && kk0 + kk + 3 < K) ? A[(long)gr * K + kk0 + kk + 3] : 0.f;
          ra[it] = t;
        }
      }
      {                                         // ---- B tile 32x128 ----
        int kk = q >> 5, c = (q & 31) * 4;
        int gk = kk0 + kk, gc = tileN + c;
        if (fb) {
          rb[it] = *reinterpret_cast<const float4*>(W + (long)gk * N + gc);
        } else {
          float4 t;
          t.x = (gk < K && gc + 0 < N) ? W[(long)gk * N + gc + 0] : 0.f;
          t.y = (gk < K && gc + 1 < N) ? W[(long)gk * N + gc + 1] : 0.f;
          t.z = (gk < K && gc + 2 < N) ? W[(long)gk * N + gc + 2] : 0.f;
          t.w = (gk < K && gc + 3 < N) ? W[(long)gk * N + gc + 3] : 0.f;
          rb[it] = t;
        }
      }
    }
  };

  load_tiles(0);  // prologue

  for (int k0 = 0; k0 < K; k0 += 32) {
    // ---- commit staged registers to LDS (bf16, swizzled) ----
    for (int it = 0; it < 4; ++it) {
      int q = it * 256 + tid;
      {
        int row = q >> 3, kk = (q & 7) * 4;
        int h = (kk >> 3) & 1, ii = (kk & 7) + ((kk >> 4) << 3);
        uint2 u;
        u.x = pack2bf(ra[it].x, ra[it].y);
        u.y = pack2bf(ra[it].z, ra[it].w);
        *reinterpret_cast<uint2*>(&As[row * 32 + h * 16 + ii]) = u;  // 8B aligned
      }
      {
        int kk = q >> 5, c = (q & 31) * 4;
        int bo = ((kk >> 3) & 1) * 16 + (kk & 7) + ((kk >> 4) << 3);
        Bs[(c + 0) * 32 + bo] = f2bf(rb[it].x);
        Bs[(c + 1) * 32 + bo] = f2bf(rb[it].y);
        Bs[(c + 2) * 32 + bo] = f2bf(rb[it].z);
        Bs[(c + 3) * 32 + bo] = f2bf(rb[it].w);
      }
    }
    __syncthreads();

    // prefetch next k-step into registers (overlaps with frag loads + WMMA)
    if (k0 + 32 < K) load_tiles(k0 + 32);

    v16bf af[2], bfv[4];
    for (int i = 0; i < 2; ++i) {
      int r = waveM * 32 + i * 16 + l16;
      af[i] = *reinterpret_cast<const v16bf*>(&As[r * 32 + half * 16]);
    }
    for (int j = 0; j < 4; ++j) {
      int c = waveN * 64 + j * 16 + l16;
      bfv[j] = *reinterpret_cast<const v16bf*>(&Bs[c * 32 + half * 16]);
    }
    for (int i = 0; i < 2; ++i)
      for (int j = 0; j < 4; ++j)
        acc[i][j] = __builtin_amdgcn_wmma_f32_16x16x32_bf16(
            false, af[i], false, bfv[j], (short)0, acc[i][j], false, false);
    __syncthreads();
  }

  // ---- epilogue ----
  for (int i = 0; i < 2; ++i) {
    for (int j = 0; j < 4; ++j) {
      int col = tileN + waveN * 64 + j * 16 + l16;
      if (col >= N) continue;
      float bv = bias ? bias[col] : 0.f;
      for (int r = 0; r < 8; ++r) {
        int row = tileM + waveM * 32 + i * 16 + r + half * 8;
        if (row >= M) continue;
        long idx = (long)row * N + col;
        float s = acc[i][j][r] + bv;
        if (addBeforeAct && D) s += D[idx];
        s = act_apply(s, act);
        if (!addBeforeAct && D) s += D[idx];
        C[idx] = s;
      }
    }
  }
}

// ---------------------------------------------------------------------------
// small / memory-bound kernels
// ---------------------------------------------------------------------------
__global__ void fill_u32_kernel(unsigned* p, unsigned v, int n) {
  int i = blockIdx.x * blockDim.x + threadIdx.x;
  if (i < n) p[i] = v;
}

__global__ void deg_kernel(const int* __restrict__ src, float* deg, int E) {
  int e = blockIdx.x * blockDim.x + threadIdx.x;
  if (e < E) atomicAdd(deg + src[e], 1.0f);
}

__global__ __launch_bounds__(256) void l2norm_rows_kernel(float* X, int rows, int L) {
  int row  = (blockIdx.x * blockDim.x + threadIdx.x) >> 5;
  int lane = threadIdx.x & 31;
  if (row >= rows) return;
  float* x = X + (size_t)row * L;
  float ss = 0.f;
  for (int l = lane; l < L; l += 32) { float v = x[l]; ss += v * v; }
  for (int o = 16; o > 0; o >>= 1) ss += __shfl_down(ss, o, 32);
  ss = __shfl(ss, 0, 32);
  float scale = 1.f / fmaxf(sqrtf(ss), 1e-12f);
  for (int l = lane; l < L; l += 32) x[l] *= scale;
}

// per-edge gated attention logit + segment max (atomic float-max bit trick)
__global__ __launch_bounds__(256) void edge_logits_kernel(
    const float* __restrict__ XC, const int* __restrict__ src,
    const int* __restrict__ dst, const float* __restrict__ deg,
    float* __restrict__ logits, float* __restrict__ mbuf, int E, int L)
{
  int e    = (blockIdx.x * blockDim.x + threadIdx.x) >> 5;
  int lane = threadIdx.x & 31;
  if (e >= E) return;
  int s = src[e], d = dst[e];
  const float* xj = XC + (size_t)s * L;
  const float* xi = XC + (size_t)d * L;
  float acc = 0.f;
  for (int l = lane; l < L; l += 32) {
    float a = xj[l];
    a = a > 0.f ? a : 0.01f * a;           // leaky_relu(x_j)
    acc += xi[l] * a;
  }
  for (int o = 16; o > 0; o >>= 1) acc += __shfl_down(acc, o, 32);
  if (lane == 0) {
    float inner = acc;
    float gate  = 1.f / (1.f + expf(-inner * rsqrtf(deg[s])));
    float lg    = inner * gate;
    logits[e] = lg;
    if (lg >= 0.f) atomicMax((int*)(mbuf + d), __float_as_int(lg));
    else           atomicMin((unsigned int*)(mbuf + d), __float_as_uint(lg));
  }
}

__global__ void edge_expz_kernel(const float* __restrict__ logits,
                                 const int* __restrict__ dst,
                                 const float* __restrict__ mbuf,
                                 float* __restrict__ ea, float* __restrict__ zbuf, int E)
{
  int e = blockIdx.x * blockDim.x + threadIdx.x;
  if (e >= E) return;
  int d = dst[e];
  float v = expf(logits[e] - mbuf[d]);
  ea[e] = v;
  atomicAdd(zbuf + d, v);
}

__global__ __launch_bounds__(256) void edge_scatter_kernel(
    const float* __restrict__ XC, const int* __restrict__ src,
    const int* __restrict__ dst, const float* __restrict__ ea,
    const float* __restrict__ zbuf, float* __restrict__ out, int E, int L)
{
  int e    = (blockIdx.x * blockDim.x + threadIdx.x) >> 5;
  int lane = threadIdx.x & 31;
  if (e >= E) return;
  int s = src[e], d = dst[e];
  float alpha = ea[e] / zbuf[d];
  const float* xj = XC + (size_t)s * L;
  float* o = out + (size_t)d * L;
  for (int l = lane; l < L; l += 32) atomicAdd(o + l, alpha * xj[l]);
}

// h = leaky_relu(l2norm(out + conv_b)) in place
__global__ __launch_bounds__(256) void post_gat_kernel(
    float* __restrict__ OUT, const float* __restrict__ bias, int rows, int L)
{
  int row  = (blockIdx.x * blockDim.x + threadIdx.x) >> 5;
  int lane = threadIdx.x & 31;
  if (row >= rows) return;
  float* o = OUT + (size_t)row * L;
  float ss = 0.f;
  for (int l = lane; l < L; l += 32) { float v = o[l] + bias[l]; ss += v * v; }
  for (int s = 16; s > 0; s >>= 1) ss += __shfl_down(ss, s, 32);
  ss = __shfl(ss, 0, 32);
  float scale = 1.f / fmaxf(sqrtf(ss), 1e-12f);
  for (int l = lane; l < L; l += 32) {
    float v = (o[l] + bias[l]) * scale;
    o[l] = v > 0.f ? v : 0.01f * v;
  }
}

__global__ void add_kernel(float* dst, const float* __restrict__ s, int n) {
  int i = blockIdx.x * blockDim.x + threadIdx.x;
  if (i < n) dst[i] += s[i];
}

__global__ void scale_copy_kernel(float* dst, const float* __restrict__ s, float f, int n) {
  int i = blockIdx.x * blockDim.x + threadIdx.x;
  if (i < n) dst[i] = s[i] * f;
}

__global__ void gather_rows_kernel(float* dst, const float* __restrict__ src,
                                   const int* __restrict__ idx, int rows, int cols) {
  int i = blockIdx.x * blockDim.x + threadIdx.x;
  if (i < rows * cols) dst[i] = src[(size_t)idx[i / cols] * cols + (i % cols)];
}

// K row m = mean_b rep[pos[b]]
__global__ void kmat_kernel(float* krow, const float* __restrict__ rep,
                            const int* __restrict__ pos, int Bn) {
  int j = threadIdx.x;  // 64
  float s = 0.f;
  for (int b = 0; b < Bn; ++b) s += rep[(size_t)pos[b] * 64 + j];
  krow[j] = s / (float)Bn;
}

__global__ void kqv_kernel(const float* __restrict__ K4, const float* __restrict__ kw,
                           const float* __restrict__ vw, float* Kp, float* Vp) {
  int j = threadIdx.x;  // 64
  for (int r = 0; r < 4; ++r) {
    float a = 0.f, b = 0.f;
    for (int l = 0; l < 64; ++l) {
      float kv = K4[r * 64 + l];
      a += kv * kw[l * 64 + j];
      b += kv * vw[l * 64 + j];
    }
    Kp[r * 64 + j] = a;
    Vp[r * 64 + j] = b;
  }
}

// softmax attention over 4 keys + BPR pos/neg scores
__global__ __launch_bounds__(64) void att_kernel(
    const float* __restrict__ Q, const float* __restrict__ Kp,
    const float* __restrict__ Vp, const float* __restrict__ rep,
    const int* __restrict__ pos, const int* __restrict__ neg,
    float* __restrict__ att, float* __restrict__ pos_out, float* __restrict__ neg_out)
{
  __shared__ float qrow[64], w4[4], red[64];
  int b = blockIdx.x, j = threadIdx.x;
  qrow[j] = Q[(size_t)b * 64 + j];
  __syncthreads();
  if (j == 0) {
    float sc[4];
    for (int r = 0; r < 4; ++r) {
      float s = 0.f;
      for (int l = 0; l < 64; ++l) s += qrow[l] * Kp[r * 64 + l];
      sc[r] = s * 0.125f;  // / sqrt(64)
    }
    float mx = fmaxf(fmaxf(sc[0], sc[1]), fmaxf(sc[2], sc[3]));
    float z = 0.f;
    for (int r = 0; r < 4; ++r) { sc[r] = expf(sc[r] - mx); z += sc[r]; }
    for (int r = 0; r < 4; ++r) w4[r] = sc[r] / z;
  }
  __syncthreads();
  float a = 0.f;
  for (int r = 0; r < 4; ++r) a += w4[r] * Vp[r * 64 + j];
  att[(size_t)b * 64 + j] = a;
  float pj = rep[(size_t)pos[b] * 64 + j];
  float nj = rep[(size_t)neg[b] * 64 + j];
  red[j] = a * pj; __syncthreads();
  for (int o = 32; o > 0; o >>= 1) { if (j < o) red[j] += red[j + o]; __syncthreads(); }
  if (j == 0) pos_out[b] = red[0];
  __syncthreads();
  red[j] = a * nj; __syncthreads();
  for (int o = 32; o > 0; o >>= 1) { if (j < o) red[j] += red[j + o]; __syncthreads(); }
  if (j == 0) neg_out[b] = red[0];
}

// price head: sigmoid(leaky([att,pos] @ p1 + b1) @ p2 + b2)
__global__ __launch_bounds__(64) void pred_kernel(
    const float* __restrict__ att, const float* __restrict__ rep,
    const int* __restrict__ pos, const float* __restrict__ p1w,
    const float* __restrict__ p1b, const float* __restrict__ p2w,
    const float* __restrict__ p2b, float* __restrict__ out)
{
  __shared__ float up[128], red[64];
  int b = blockIdx.x, j = threadIdx.x;
  up[j]      = att[(size_t)b * 64 + j];
  up[64 + j] = rep[(size_t)pos[b] * 64 + j];
  __syncthreads();
  float h = p1b[j];
  for (int k = 0; k < 128; ++k) h += up[k] * p1w[k * 64 + j];
  h = h > 0.f ? h : 0.01f * h;
  red[j] = h * p2w[j]; __syncthreads();
  for (int o = 32; o > 0; o >>= 1) { if (j < o) red[j] += red[j + o]; __syncthreads(); }
  if (j == 0) out[b] = 1.f / (1.f + expf(-(red[0] + p2b[0])));
}

// ---------------------------------------------------------------------------
static void launch_gemm(const float* A, const float* W, const float* bias,
                        const float* D, float* C, int M, int N, int K,
                        int act, int addBeforeAct, hipStream_t s) {
  dim3 grid((N + 127) / 128, (M + 127) / 128);
  gemm_bf16_kernel<<<grid, dim3(256), 0, s>>>(A, W, bias, D, C, M, N, K, act, addBeforeAct);
}

extern "C" void kernel_launch(void* const* d_in, const int* in_sizes, int n_in,
                              void* d_out, int out_size, void* d_ws, size_t ws_size,
                              hipStream_t stream)
{
  const int NUM_USER = 5000, NUM_ITEM = 20000, NN = 25000, DX = 64, BN = 2048, UD = 128;
  const int fdim[4] = {2048, 768, 64, 64};
  const int ldim[4] = {1024, 1500, 64, 64};

  const float* feats[4] = {(const float*)d_in[0], (const float*)d_in[1],
                           (const float*)d_in[2], (const float*)d_in[3]};
  const float* uf  = (const float*)d_in[4];
  const int*   ei  = (const int*)d_in[5];
  const int    E   = in_sizes[5] / 2;       // 100000
  const int*   src = ei;
  const int*   dst = ei + E;
  const int*   unodes = (const int*)d_in[6];
  const int*   pos    = (const int*)d_in[7];
  const int*   neg    = (const int*)d_in[8];
  // params: per modality [mlp_w,mlp_b,umlp_w,umlp_b,conv_w,conv_b,lin_w,lin_b,g_w,g_b]
  const float* Pm[4][10];
  for (int m = 0; m < 4; ++m)
    for (int j = 0; j < 10; ++j) Pm[m][j] = (const float*)d_in[9 + m * 10 + j];
  const float* id_emb = (const float*)d_in[49];
  const float* q_w  = (const float*)d_in[50];
  const float* k_w  = (const float*)d_in[51];
  const float* v_w  = (const float*)d_in[52];
  const float* p1w  = (const float*)d_in[53];
  const float* p1b  = (const float*)d_in[54];
  const float* p2w  = (const float*)d_in[55];
  const float* p2b  = (const float*)d_in[56];
  (void)n_in; (void)out_size; (void)ws_size;

  float* out_pos  = (float*)d_out;
  float* out_neg  = out_pos + BN;
  float* out_rep  = out_neg + BN;                       // [25000,64]
  float* out_pred = out_rep + (size_t)NN * DX;          // [2048]

  // workspace layout (floats)
  const size_t LMAX = 1500;
  float* ws = (float*)d_ws;
  size_t off = 0;
  float* X      = ws + off; off += (size_t)NN * LMAX;   // x, later reused as OUT/H
  float* XC     = ws + off; off += (size_t)NN * LMAX;   // conv-transformed x
  float* REPM   = ws + off; off += (size_t)NN * DX;     // per-modality rep
  float* REPSUM = ws + off; off += (size_t)NN * DX;
  float* DEG    = ws + off; off += NN;
  float* MB     = ws + off; off += NN;                  // segment max
  float* ZB     = ws + off; off += NN;                  // segment sum
  float* LOG    = ws + off; off += E;
  float* EA     = ws + off; off += E;
  float* UT     = ws + off; off += (size_t)BN * DX;
  float* QB     = ws + off; off += (size_t)BN * DX;
  float* ATT    = ws + off; off += (size_t)BN * DX;
  float* KM     = ws + off; off += 4 * DX;
  float* KP     = ws + off; off += 4 * DX;
  float* VP     = ws + off; off += 4 * DX;

  hipMemsetAsync(REPSUM, 0, (size_t)NN * DX * sizeof(float), stream);
  hipMemsetAsync(DEG, 0, NN * sizeof(float), stream);
  deg_kernel<<<(E + 255) / 256, 256, 0, stream>>>(src, DEG, E);

  for (int m = 0; m < 4; ++m) {
    const int L = ldim[m], Dd = fdim[m];
    // x = l2norm(concat(tanh(feat@mlp+b), tanh(user@umlp+b)))
    launch_gemm(feats[m], Pm[m][0], Pm[m][1], nullptr, X, NUM_ITEM, L, Dd, ACT_TANH, 0, stream);
    launch_gemm(uf, Pm[m][2], Pm[m][3], nullptr, X + (size_t)NUM_ITEM * L,
                NUM_USER, L, UD, ACT_TANH, 0, stream);
    l2norm_rows_kernel<<<(NN * 32 + 255) / 256, 256, 0, stream>>>(X, NN, L);
    // x_hat = leaky(x@lin+b) + id_emb   (needs x, so do before reusing X)
    launch_gemm(X, Pm[m][6], Pm[m][7], id_emb, REPM, NN, DX, L, ACT_LEAKY, 0, stream);
    // xc = x @ conv_w
    launch_gemm(X, Pm[m][4], nullptr, nullptr, XC, NN, L, L, ACT_NONE, 0, stream);
    // GAT edge stage (X reused as OUT accumulator)
    hipMemsetAsync(X, 0, (size_t)NN * L * sizeof(float), stream);
    fill_u32_kernel<<<(NN + 255) / 256, 256, 0, stream>>>((unsigned*)MB, 0xFF800000u, NN);
    hipMemsetAsync(ZB, 0, NN * sizeof(float), stream);
    edge_logits_kernel<<<(E * 32 + 255) / 256, 256, 0, stream>>>(XC, src, dst, DEG, LOG, MB, E, L);
    edge_expz_kernel<<<(E + 255) / 256, 256, 0, stream>>>(LOG, dst, MB, EA, ZB, E);
    edge_scatter_kernel<<<(E * 32 + 255) / 256, 256, 0, stream>>>(XC, src, dst, EA, ZB, X, E, L);
    // h = leaky(l2norm(out + conv_b))
    post_gat_kernel<<<(NN * 32 + 255) / 256, 256, 0, stream>>>(X, Pm[m][5], NN, L);
    // rep_m = leaky(h@g_w + g_b + x_hat)
    launch_gemm(X, Pm[m][8], Pm[m][9], REPM, REPM, NN, DX, L, ACT_LEAKY, 1, stream);
    add_kernel<<<(NN * DX + 255) / 256, 256, 0, stream>>>(REPSUM, REPM, NN * DX);
    kmat_kernel<<<1, 64, 0, stream>>>(KM + m * DX, REPM, pos, BN);
  }

  // representation = sum/4  -> directly into d_out
  scale_copy_kernel<<<(NN * DX + 255) / 256, 256, 0, stream>>>(out_rep, REPSUM, 0.25f, NN * DX);
  // attention tail
  gather_rows_kernel<<<(BN * DX + 255) / 256, 256, 0, stream>>>(UT, out_rep, unodes, BN, DX);
  launch_gemm(UT, q_w, nullptr, nullptr, QB, BN, DX, DX, ACT_NONE, 0, stream);
  kqv_kernel<<<1, 64, 0, stream>>>(KM, k_w, v_w, KP, VP);
  att_kernel<<<BN, 64, 0, stream>>>(QB, KP, VP, out_rep, pos, neg, ATT, out_pos, out_neg);
  pred_kernel<<<BN, 64, 0, stream>>>(ATT, out_rep, pos, p1w, p1b, p2w, p2b, out_pred);
}
